// PointCloudAE_21139829031414
// MI455X (gfx1250) — matrix operand order repaired and checked by reading, not compiled, not run on hardware
//
#include <hip/hip_runtime.h>
#include <hip/hip_bf16.h>
#include <stdint.h>

// ---------------- problem constants ----------------
#define B_   8
#define N_   16384
#define K_   16
#define M1_  1024
#define R0_  0.2f
#define R1_  0.5f
// dims: D0=128 H0=64 H1=256 D1=512 HD0=256 HD1=128
// padded K dims for WMMA (multiples of 32): 131->160, 514->544, 130->160

typedef _Float16 h16;
typedef __attribute__((ext_vector_type(16))) _Float16 v16h;
typedef __attribute__((ext_vector_type(8)))  _Float16 v8h;
typedef __attribute__((ext_vector_type(8)))  float    v8f;

// ---------------- WMMA helpers (CDNA5, wave32) ----------------
static __device__ __forceinline__ v8f wmma32(v16h a, v16h b, v8f c) {
  // D(16x16,f32) = A(16x32,f16) * B(32x16,f16) + C
  return __builtin_amdgcn_wmma_f32_16x16x32_f16(false, a, false, b, (short)0, c,
                                                false, false);
}

// A fragment: row = m0 + (lane&15); lanes<16 hold K {kb+0..7, kb+16..23},
// lanes>=16 hold K {kb+8..15, kb+24..31}  (per ISA 16-bit A 16x32 layout).
static __device__ __forceinline__ v16h fragA_load(const h16* __restrict__ A, int lda,
                                                  int m0, int kb, int lane) {
  const h16* p = A + (size_t)(m0 + (lane & 15)) * lda + kb + ((lane & 16) ? 8 : 0);
  v8h lo = *(const v8h*)p;
  v8h hi = *(const v8h*)(p + 16);
  v16h r;
#pragma unroll
  for (int i = 0; i < 8; ++i) { r[i] = lo[i]; r[8 + i] = hi[i]; }
  return r;
}

// B fragment from N-major (transposed) weights WT[N][Kd]:
// col n = n0 + (lane&15); lanes<16 hold K {kb+0..7, kb+16..23},
// lanes>=16 hold K {kb+8..15, kb+24..31} (same order as A).
static __device__ __forceinline__ v16h fragB_load(const h16* __restrict__ WT, int ldw,
                                                  int n0, int kb, int lane) {
  const h16* p = WT + (size_t)(n0 + (lane & 15)) * ldw + kb + ((lane & 16) ? 8 : 0);
  v8h lo = *(const v8h*)p;
  v8h hi = *(const v8h*)(p + 16);
  v16h r;
#pragma unroll
  for (int i = 0; i < 8; ++i) { r[i] = lo[i]; r[8 + i] = hi[i]; }
  return r;
}

// ---------------- 1) weight prep: f16 + transpose + zero-pad ----------------
__global__ void k_prep(const float* __restrict__ W0b,  const float* __restrict__ W1a,
                       const float* __restrict__ W1b,  const float* __restrict__ Wd0a,
                       const float* __restrict__ Wf0,  const float* __restrict__ Wd1a,
                       h16* w0bt, h16* w1at, h16* w1bt, h16* wd0at, h16* wf0t, h16* wd1at) {
  int t = blockIdx.x * 256 + threadIdx.x;
  if (t < 8192)  { int n = t / 64,  k = t % 64;  w0bt[t]  = (h16)W0b[k * 128 + n]; return; }
  t -= 8192;
  if (t < 40960) { int n = t / 160, k = t % 160; w1at[t]  = (h16)(k < 131 ? W1a[k * 256 + n] : 0.f); return; }
  t -= 40960;
  if (t < 131072){ int n = t / 256, k = t % 256; w1bt[t]  = (h16)W1b[k * 512 + n]; return; }
  t -= 131072;
  if (t < 139264){ int n = t / 544, k = t % 544; wd0at[t] = (h16)(k < 514 ? Wd0a[k * 256 + n] : 0.f); return; }
  t -= 139264;
  if (t < 32768) { int n = t / 256, k = t % 256; wf0t[t]  = (h16)Wf0[k * 128 + n]; return; }
  t -= 32768;
  if (t < 20480) { int n = t / 160, k = t % 160; wd1at[t] = (h16)(k < 130 ? Wd1a[k * 128 + n] : 0.f); return; }
}

// ---------------- 2) FPS: one 1024-thread block per cloud, register-resident ----------------
__global__ void __launch_bounds__(1024) k_fps(const float* __restrict__ pts,
                                              int* __restrict__ fidx, float* __restrict__ p1,
                                              h16* __restrict__ x1, float* __restrict__ out_ip1) {
  __shared__ float swv[32];
  __shared__ int   swi[32];
  __shared__ int   s_next;
  __shared__ int   s_fidx[M1_];
  const int b = blockIdx.x, tid = threadIdx.x;
  const int lane = tid & 31, wave = tid >> 5;
  const float* pb = pts + (size_t)b * N_ * 3;

  float px[16], py[16], pz[16], d[16];
  const float sx = pb[0], sy = pb[1], sz = pb[2];
#pragma unroll
  for (int j = 0; j < 16; ++j) {
    const int i = j * 1024 + tid;
    const float* q = pb + (size_t)i * 3;
    px[j] = q[0]; py[j] = q[1]; pz[j] = q[2];
    const float dx = px[j] - sx, dy = py[j] - sy, dz = pz[j] - sz;
    d[j] = dx * dx + dy * dy + dz * dz;
  }
  if (tid == 0) s_fidx[0] = 0;

  for (int it = 1; it < M1_; ++it) {
    // local argmax over this thread's 16 points
    float bv = d[0]; int bj = 0;
#pragma unroll
    for (int j = 1; j < 16; ++j) if (d[j] > bv) { bv = d[j]; bj = j; }
    int bidx = bj * 1024 + tid;
    // wave32 reduce (prefer lower index on ties)
#pragma unroll
    for (int off = 16; off > 0; off >>= 1) {
      const float ov = __shfl_xor(bv, off, 32);
      const int   oi = __shfl_xor(bidx, off, 32);
      if (ov > bv || (ov == bv && oi < bidx)) { bv = ov; bidx = oi; }
    }
    if (lane == 0) { swv[wave] = bv; swi[wave] = bidx; }
    __syncthreads();
    if (wave == 0) {
      float v = swv[lane]; int ii = swi[lane];
#pragma unroll
      for (int off = 16; off > 0; off >>= 1) {
        const float ov = __shfl_xor(v, off, 32);
        const int   oi = __shfl_xor(ii, off, 32);
        if (ov > v || (ov == v && oi < ii)) { v = ov; ii = oi; }
      }
      if (lane == 0) { s_next = ii; s_fidx[it] = ii; }
    }
    __syncthreads();
    const int nxt = s_next;
    const float* q = pb + (size_t)nxt * 3;
    const float nx = q[0], ny = q[1], nz = q[2];
#pragma unroll
    for (int j = 0; j < 16; ++j) {
      const float dx = px[j] - nx, dy = py[j] - ny, dz = pz[j] - nz;
      d[j] = fminf(d[j], dx * dx + dy * dy + dz * dz);
    }
  }
  __syncthreads();
  // finalize: thread m handles anchor m
  const int m = tid;
  const int idx = s_fidx[m];
  const size_t pm = (size_t)b * M1_ + m;
  fidx[pm] = idx;
  const float* q = pb + (size_t)idx * 3;
  const float qx = q[0], qy = q[1], qz = q[2];
  p1[pm * 3 + 0] = qx; p1[pm * 3 + 1] = qy; p1[pm * 3 + 2] = qz;
  out_ip1[pm * 3 + 0] = qx; out_ip1[pm * 3 + 1] = qy; out_ip1[pm * 3 + 2] = qz;
  h16* xr = x1 + pm * 160;
  xr[0] = (h16)(qx / R1_); xr[1] = (h16)(qy / R1_); xr[2] = (h16)(qz / R1_);
  for (int c = 131; c < 160; ++c) xr[c] = (h16)0.0f;   // K-padding must be finite
}

// ---------------- 3) KNN top-16 per anchor (LDS-tiled scan) ----------------
__global__ void __launch_bounds__(128) k_knn(const float* __restrict__ pts,
                                             const float* __restrict__ p1,
                                             int* __restrict__ knn, float* __restrict__ out_ip0) {
  __shared__ float sx[1024], sy[1024], sz[1024];
  const int b = blockIdx.x >> 3;
  const int a = (blockIdx.x & 7) * 128 + threadIdx.x;   // anchor 0..1023
  const float* pb = pts + (size_t)b * N_ * 3;
  const float* pa = p1 + ((size_t)b * M1_ + a) * 3;
  const float ax = pa[0], ay = pa[1], az = pa[2];
  float bd[16]; int bi[16];
#pragma unroll
  for (int j = 0; j < 16; ++j) { bd[j] = 3.4e38f; bi[j] = 0; }

  for (int t = 0; t < N_ / 1024; ++t) {
    for (int i = threadIdx.x; i < 1024; i += 128) {
      const float* q = pb + (size_t)(t * 1024 + i) * 3;
      sx[i] = q[0]; sy[i] = q[1]; sz[i] = q[2];
    }
    __syncthreads();
    for (int i = 0; i < 1024; ++i) {
      const float dx = sx[i] - ax, dy = sy[i] - ay, dz = sz[i] - az;
      const float dd = dx * dx + dy * dy + dz * dz;
      if (dd < bd[15]) {                 // insert, single bubble pass (stable)
        bd[15] = dd; bi[15] = t * 1024 + i;
#pragma unroll
        for (int j = 15; j > 0; --j) {
          if (bd[j] < bd[j - 1]) {
            const float td = bd[j]; bd[j] = bd[j - 1]; bd[j - 1] = td;
            const int   ti = bi[j]; bi[j] = bi[j - 1]; bi[j - 1] = ti;
          }
        }
      }
    }
    __syncthreads();
  }
  const size_t base = ((size_t)b * M1_ + a) * 16;
#pragma unroll
  for (int k = 0; k < 16; ++k) {
    knn[base + k] = bi[k];
    const float* q = pb + (size_t)bi[k] * 3;
    float* o = out_ip0 + ((size_t)b * N_ + a * 16 + k) * 3;
    o[0] = q[0]; o[1] = q[1]; o[2] = q[2];
  }
}

// ---------------- 4) enc0: fused rel0 -> relu(W0a)+b0a -> WMMA(W0b) -> max over K ----------------
__global__ void __launch_bounds__(256) k_enc0(const float* __restrict__ pts,
                                              const float* __restrict__ p1,
                                              const int* __restrict__ knn,
                                              const float* __restrict__ W0a,
                                              const float* __restrict__ b0a,
                                              const h16* __restrict__ W0bT,
                                              const float* __restrict__ b0b,
                                              h16* __restrict__ x1) {
  const int wave = threadIdx.x >> 5, lane = threadIdx.x & 31;
  const int bm = blockIdx.x;              // b*M1 + m
  const int b = bm >> 10;
  const int row = lane & 15;              // neighbor k (tile row)
  const int idx = knn[(size_t)bm * 16 + row];
  const float* pp = pts + ((size_t)b * N_ + idx) * 3;
  const float* pa = p1 + (size_t)bm * 3;
  const float dx = pp[0] - pa[0], dy = pp[1] - pa[1], dz = pp[2] - pa[2];
  const float d2 = dx * dx + dy * dy + dz * dz;
  float rx = 0.f, ry = 0.f, rz = 0.f;
  if (d2 <= R0_ * R0_) { rx = dx / R0_; ry = dy / R0_; rz = dz / R0_; }

  v8f acc = {};
#pragma unroll
  for (int c = 0; c < 2; ++c) {           // K=64 in two chunks of 32
    v16h a;
#pragma unroll
    for (int i = 0; i < 16; ++i) {        // layer-a in registers, A-layout slots
      const int k = c * 32 + ((lane & 16) ? 8 : 0) + (i < 8 ? i : i + 8);
      const float h = rx * W0a[k] + ry * W0a[64 + k] + rz * W0a[128 + k] + b0a[k];
      a[i] = (h16)fmaxf(h, 0.0f);
    }
    acc = wmma32(a, fragB_load(W0bT, 64, wave * 16, c * 32, lane), acc);
  }
  // max over the 16 tile rows == max over K neighbors
  const int n = wave * 16 + (lane & 15);
  float mx = acc[0];
#pragma unroll
  for (int j = 1; j < 8; ++j) mx = fmaxf(mx, acc[j]);
  mx = fmaxf(mx, __shfl_xor(mx, 16, 32));
  if (lane < 16) x1[(size_t)bm * 160 + 3 + n] = (h16)(mx + b0b[n]);  // f0 -> x1 cols 3..130
}

// ---------------- 5) generic f16 GEMM, compile-time K, 2 M-tiles/wave ----------------
template <int KD, int RELU>
__global__ void __launch_bounds__(256) k_gemm16(const h16* __restrict__ A, int lda,
                                                const h16* __restrict__ WT,
                                                const float* __restrict__ bias,
                                                h16* __restrict__ O, int ldo) {
  const int wave = threadIdx.x >> 5, lane = threadIdx.x & 31;
  const int m0 = blockIdx.x * 32;                     // two 16-row tiles per wave
  const int n0 = (blockIdx.y * 8 + wave) * 16;
  __builtin_prefetch((const void*)(A + (size_t)m0 * lda), 0, 1);
  v8f acc0 = {}, acc1 = {};
#pragma unroll
  for (int kb = 0; kb < KD; kb += 32) {               // fully unrolled
    const v16h bf = fragB_load(WT, KD, n0, kb, lane); // shared by both tiles
    acc0 = wmma32(fragA_load(A, lda, m0, kb, lane), bf, acc0);
    acc1 = wmma32(fragA_load(A, lda, m0 + 16, kb, lane), bf, acc1);
  }
  const int n = n0 + (lane & 15);
  const float bb = bias ? bias[n] : 0.0f;
  const int rbase = m0 + ((lane & 16) ? 8 : 0);
#pragma unroll
  for (int j = 0; j < 8; ++j) {
    float v0 = acc0[j] + bb;
    float v1 = acc1[j] + bb;
    if (RELU) { v0 = fmaxf(v0, 0.0f); v1 = fmaxf(v1, 0.0f); }
    O[(size_t)(rbase + j) * ldo + n] = (h16)v0;
    O[(size_t)(rbase + 16 + j) * ldo + n] = (h16)v1;
  }
}

// ---------------- 6) GEMM with fused per-tile row-max (enc1 layer b) ----------------
template <int KD>
__global__ void __launch_bounds__(256) k_gemm16_rowmax(const h16* __restrict__ A, int lda,
                                                       const h16* __restrict__ WT,
                                                       float* __restrict__ P, int Nld) {
  const int wave = threadIdx.x >> 5, lane = threadIdx.x & 31;
  const int m0 = blockIdx.x * 32;                     // two 16-row tiles per wave
  const int n0 = (blockIdx.y * 8 + wave) * 16;
  v8f acc0 = {}, acc1 = {};
#pragma unroll
  for (int kb = 0; kb < KD; kb += 32) {
    const v16h bf = fragB_load(WT, KD, n0, kb, lane);
    acc0 = wmma32(fragA_load(A, lda, m0, kb, lane), bf, acc0);
    acc1 = wmma32(fragA_load(A, lda, m0 + 16, kb, lane), bf, acc1);
  }
  float mx0 = acc0[0], mx1 = acc1[0];
#pragma unroll
  for (int j = 1; j < 8; ++j) { mx0 = fmaxf(mx0, acc0[j]); mx1 = fmaxf(mx1, acc1[j]); }
  mx0 = fmaxf(mx0, __shfl_xor(mx0, 16, 32));
  mx1 = fmaxf(mx1, __shfl_xor(mx1, 16, 32));
  if (lane < 16) {
    P[(size_t)(blockIdx.x * 2 + 0) * Nld + n0 + lane] = mx0;
    P[(size_t)(blockIdx.x * 2 + 1) * Nld + n0 + lane] = mx1;
  }
}

// ---------------- 7) z = max over 64 partial row-tiles + b1b ----------------
__global__ void k_zreduce(const float* __restrict__ P, const float* __restrict__ b1b,
                          float* __restrict__ z, float* __restrict__ zout) {
  const int t = blockIdx.x * blockDim.x + threadIdx.x;
  if (t >= B_ * 512) return;
  const int b = t >> 9, n = t & 511;
  float m = -3.4e38f;
  for (int i = 0; i < 64; ++i) m = fmaxf(m, P[(size_t)(b * 64 + i) * 512 + n]);
  const float v = m + b1b[n];
  z[t] = v; zout[t] = v;
}

// ---------------- 8) build decoder-0 input: [z | grid0 | pad] in f16 ----------------
__global__ void k_xd0(const float* __restrict__ z, const float* __restrict__ grid0,
                      h16* __restrict__ xd0) {
  const int t = blockIdx.x * blockDim.x + threadIdx.x;
  if (t >= B_ * M1_ * 544) return;
  const int row = t / 544, col = t % 544;
  const int b = row >> 10, m = row & 1023;
  float v;
  if (col < 512)      v = z[b * 512 + col];
  else if (col < 514) v = grid0[m * 2 + (col - 512)];
  else                v = 0.0f;
  xd0[t] = (h16)v;
}

// ---------------- 9) relp0 = hd0 @ Wp0 ; out1 = relp0 * R1 ----------------
__global__ void k_relp0(const h16* __restrict__ hd0, const float* __restrict__ Wp0,
                        float* __restrict__ out1) {
  const int t = blockIdx.x * blockDim.x + threadIdx.x;
  if (t >= B_ * M1_ * 3) return;
  const int row = t / 3, j = t % 3;
  const h16* hr = hd0 + (size_t)row * 256;
  float s = 0.0f;
  for (int k = 0; k < 256; ++k) s += (float)hr[k] * Wp0[k * 3 + j];
  out1[t] = s * R1_;
}

// ---------------- 10) decoder-1 GEMM: A tile = [fdec[m] bcast | grid1 | 0] ----------------
__global__ void __launch_bounds__(256) k_dec1(const h16* __restrict__ fdec,
                                              const float* __restrict__ grid1,
                                              const h16* __restrict__ WT,     // [128][160]
                                              const float* __restrict__ bd1a,
                                              h16* __restrict__ hd1) {
  const int wave = threadIdx.x >> 5, lane = threadIdx.x & 31;
  const int rowg = blockIdx.x;            // b*M1 + m  (one 16-row tile)
  const h16* fr = fdec + (size_t)rowg * 128;
  const int n0 = wave * 16;
  v8f acc = {};
#pragma unroll
  for (int c = 0; c < 4; ++c) {           // K 0..127 from fdec (same for all 16 rows)
    const int kb = c * 32;
    const h16* p = fr + kb + ((lane & 16) ? 8 : 0);
    v8h lo = *(const v8h*)p;
    v8h hi = *(const v8h*)(p + 16);
    v16h a;
#pragma unroll
    for (int i = 0; i < 8; ++i) { a[i] = lo[i]; a[8 + i] = hi[i]; }
    acc = wmma32(a, fragB_load(WT, 160, n0, kb, lane), acc);
  }
  {                                       // K 128..159: grid1 row per tile-row, rest 0
    v16h a = {};
    if (!(lane & 16)) {
      const int k = lane & 15;            // tile row == neighbor index
      a[0] = (h16)grid1[k * 2 + 0];
      a[1] = (h16)grid1[k * 2 + 1];
    }
    acc = wmma32(a, fragB_load(WT, 160, n0, 128, lane), acc);
  }
  const int n = n0 + (lane & 15);
  const float bb = bd1a[n];
  const size_t rb = (size_t)rowg * 16 + ((lane & 16) ? 8 : 0);
#pragma unroll
  for (int j = 0; j < 8; ++j)
    hd1[(rb + j) * 128 + n] = (h16)fmaxf(acc[j] + bb, 0.0f);
}

// ---------------- 11) out0 = out1[m] + (hd1 @ Wp1) * R0 ----------------
__global__ void k_final(const h16* __restrict__ hd1, const float* __restrict__ Wp1,
                        const float* __restrict__ out1, float* __restrict__ out0) {
  const int r = blockIdx.x * blockDim.x + threadIdx.x;
  if (r >= B_ * N_) return;
  const int b = r >> 14, mrem = r & 16383, m = mrem >> 4;
  const h16* hr = hd1 + (size_t)r * 128;
  float s0 = 0.f, s1 = 0.f, s2 = 0.f;
  for (int k = 0; k < 128; ++k) {
    const float h = (float)hr[k];
    s0 += h * Wp1[k * 3 + 0]; s1 += h * Wp1[k * 3 + 1]; s2 += h * Wp1[k * 3 + 2];
  }
  const float* o1 = out1 + ((size_t)b * M1_ + m) * 3;
  out0[(size_t)r * 3 + 0] = o1[0] + s0 * R0_;
  out0[(size_t)r * 3 + 1] = o1[1] + s1 * R0_;
  out0[(size_t)r * 3 + 2] = o1[2] + s2 * R0_;
}

// ---------------- host-side launcher ----------------
extern "C" void kernel_launch(void* const* d_in, const int* in_sizes, int n_in,
                              void* d_out, int out_size, void* d_ws, size_t ws_size,
                              hipStream_t stream) {
  (void)in_sizes; (void)n_in; (void)out_size; (void)ws_size;
  const float* points = (const float*)d_in[0];
  const float* W0a  = (const float*)d_in[1];
  const float* b0a  = (const float*)d_in[2];
  const float* W0b  = (const float*)d_in[3];
  const float* b0b  = (const float*)d_in[4];
  const float* W1a  = (const float*)d_in[5];
  const float* b1a  = (const float*)d_in[6];
  const float* W1b  = (const float*)d_in[7];
  const float* b1b  = (const float*)d_in[8];
  const float* grid0= (const float*)d_in[9];
  const float* Wd0a = (const float*)d_in[10];
  const float* bd0a = (const float*)d_in[11];
  const float* Wp0  = (const float*)d_in[12];
  const float* Wf0  = (const float*)d_in[13];
  const float* grid1= (const float*)d_in[14];
  const float* Wd1a = (const float*)d_in[15];
  const float* bd1a = (const float*)d_in[16];
  const float* Wp1  = (const float*)d_in[17];

  // d_out sections (floats): ip0 | ip1 | out0 | out1 | z
  float* out   = (float*)d_out;
  float* o_ip0 = out;
  float* o_ip1 = out + 393216;
  float* o_out0= out + 417792;
  float* o_out1= out + 811008;
  float* o_z   = out + 835584;

  // workspace layout (bytes; all 256-aligned)
  char* ws = (char*)d_ws;
  int*   fidx  = (int*)  (ws + 0);          // 32768
  float* p1    = (float*)(ws + 32768);      // 98304
  int*   knn   = (int*)  (ws + 131072);     // 524288
  h16*   x1    = (h16*)  (ws + 655360);     // [B*1024][160]
  h16*   h1    = (h16*)  (ws + 3276800);    // [B*1024][256]
  float* part  = (float*)(ws + 7471104);    // [512][512]
  float* z     = (float*)(ws + 8519680);    // [B][512]
  h16*   xd0   = (h16*)  (ws + 8536064);    // [B*1024][544]
  h16*   hd0   = (h16*)  (ws + 17448960);   // [B*1024][256]
  h16*   fdec  = (h16*)  (ws + 21643264);   // [B*1024][128]
  h16*   hd1   = (h16*)  (ws + 23740416);   // [B*16384][128]
  h16*   w0bt  = (h16*)  (ws + 57294848);   // [128][64]
  h16*   w1at  = (h16*)  (ws + 57311232);   // [256][160]
  h16*   w1bt  = (h16*)  (ws + 57393152);   // [512][256]
  h16*   wd0at = (h16*)  (ws + 57655296);   // [256][544]
  h16*   wf0t  = (h16*)  (ws + 57933824);   // [128][256]
  h16*   wd1at = (h16*)  (ws + 57999360);   // [128][160]

  // 1) weights -> f16, transposed, padded
  k_prep<<<(372736 + 255) / 256, 256, 0, stream>>>(W0b, W1a, W1b, Wd0a, Wf0, Wd1a,
                                                   w0bt, w1at, w1bt, wd0at, wf0t, wd1at);
  // 2) FPS (latency-critical serial chain; 1 block per cloud)
  k_fps<<<B_, 1024, 0, stream>>>(points, fidx, p1, x1, o_ip1);
  // 3) KNN + ip0
  k_knn<<<B_ * 8, 128, 0, stream>>>(points, p1, knn, o_ip0);
  // 4) enc0 fused (writes f0 into x1 cols 3..130)
  k_enc0<<<B_ * M1_, 256, 0, stream>>>(points, p1, knn, W0a, b0a, w0bt, b0b, x1);
  // 5) enc1 layer a: [8192,160] x [160,256] -> h1 (relu)
  k_gemm16<160, 1><<<dim3(256, 2), 256, 0, stream>>>(x1, 160, w1at, b1a, h1, 256);
  // 6) enc1 layer b with fused row-max: [8192,256] x [256,512] -> partial [512][512]
  k_gemm16_rowmax<256><<<dim3(256, 4), 256, 0, stream>>>(h1, 256, w1bt, part, 512);
  // 7) z
  k_zreduce<<<(B_ * 512 + 255) / 256, 256, 0, stream>>>(part, b1b, z, o_z);
  // 8) decoder-0 input
  k_xd0<<<(B_ * M1_ * 544 + 255) / 256, 256, 0, stream>>>(z, grid0, xd0);
  // 9) dec0 layer a: [8192,544] x [544,256] -> hd0 (relu)
  k_gemm16<544, 1><<<dim3(256, 2), 256, 0, stream>>>(xd0, 544, wd0at, bd0a, hd0, 256);
  // 10) fdec = hd0 @ Wf0 : [8192,256] x [256,128]
  k_gemm16<256, 0><<<dim3(256, 1), 256, 0, stream>>>(hd0, 256, wf0t,
                                                     (const float*)nullptr, fdec, 128);
  // 11) out1 = (hd0 @ Wp0) * R1
  k_relp0<<<(B_ * M1_ * 3 + 255) / 256, 256, 0, stream>>>(hd0, Wp0, o_out1);
  // 12) dec1 layer a: per-anchor broadcast tiles -> hd1 (relu)
  k_dec1<<<B_ * M1_, 256, 0, stream>>>(fdec, grid1, wd1at, bd1a, hd1);
  // 13) out0 = out1[m] + (hd1 @ Wp1) * R0
  k_final<<<(B_ * N_ + 255) / 256, 256, 0, stream>>>(hd1, Wp1, o_out1, o_out0);
}
